// MultiHeadAttention_40029095199060
// MI455X (gfx1250) — matrix operand order repaired and checked by reading, not compiled, not run on hardware
//
#include <hip/hip_runtime.h>
#include <hip/hip_bf16.h>

// CDNA5 (gfx1250, wave32) fused multi-head attention.
// All matmuls use V_WMMA_F32_16X16X32_F16. Fragments are loaded directly from
// global memory in the CDNA5 WMMA per-lane layout (CDNA5 ISA 7.12.2):
//   A: lane l, half i -> M = l%16, K = (i/8)*16 + (l/16)*8 + (i%8)
//      => per lane: two contiguous 8-half runs at k0+lg*8 and k0+16+lg*8
//   B: lane l, half i -> N = l%16, K = (i/8)*16 + (l/16)*8 + (i%8)
//      => per lane: two contiguous 8-half runs at k0+lg*8 and k0+16+lg*8
//   D: vgpr r, lane l -> M = r + 8*(l/16), N = l%16
// The fused kernel additionally stages the Q tile through LDS with the
// Tensor Data Mover (TENSOR_LOAD_TO_LDS + S_WAIT_TENSORCNT).

typedef __attribute__((ext_vector_type(16))) _Float16 v16h;
typedef __attribute__((ext_vector_type(8)))  _Float16 v8h;
typedef __attribute__((ext_vector_type(8)))  float    v8f;
typedef __attribute__((ext_vector_type(4)))  unsigned int v4u;
typedef __attribute__((ext_vector_type(8)))  int      v8i;
typedef __attribute__((ext_vector_type(4)))  int      v4i;

__device__ __forceinline__ v8f wmma_f16(v16h a, v16h b, v8f c) {
  return __builtin_amdgcn_wmma_f32_16x16x32_f16(false, a, false, b, (short)0,
                                                c, false, false);
}

// A-fragment from a row-major row pointer (convert to f16 if needed).
template <typename T>
__device__ __forceinline__ v16h load_a_frag(const T* __restrict__ row, int k0,
                                            int lg) {
  const T* p = row + k0 + lg * 8;
  v16h r;
#pragma unroll
  for (int i = 0; i < 8; ++i) r[i] = (_Float16)p[i];
#pragma unroll
  for (int i = 0; i < 8; ++i) r[8 + i] = (_Float16)p[16 + i];
  return r;
}

// B-fragment (B stored N x K row-major, i.e. A@B^T) from a row pointer.
template <typename T>
__device__ __forceinline__ v16h load_b_frag(const T* __restrict__ row, int k0,
                                            int lg) {
  const T* p = row + k0 + lg * 8;
  v16h r;
#pragma unroll
  for (int i = 0; i < 8; ++i) r[i] = (_Float16)p[i];
#pragma unroll
  for (int i = 0; i < 8; ++i) r[8 + i] = (_Float16)p[16 + i];
  return r;
}

// ---------------------------------------------------------------------------
// Generic batched C[M,N] = scale * A[M,K] @ B[N,K]^T, direct global fragments.
// Block = 256 threads = 8 waves; wave w owns m-tile (blockIdx.x*8 + w),
// all waves in a block share one n-tile (B fragment reuse through cache).
// M % 128 == 0, N % 16 == 0, K % 32 == 0.
// ---------------------------------------------------------------------------
template <typename TA, typename TB, typename TC>
__global__ __launch_bounds__(256) void wmma_gemm_abt(
    const TA* __restrict__ A, const TB* __restrict__ B, TC* __restrict__ C,
    int Kd, long long lda, long long ldb, long long ldc, long long sA,
    long long sB, long long sC, float scale) {
  const int w = threadIdx.x >> 5;
  const int lane = threadIdx.x & 31;
  const int lr = lane & 15;
  const int lg = lane >> 4;

  A += (long long)blockIdx.z * sA;
  B += (long long)blockIdx.z * sB;
  C += (long long)blockIdx.z * sC;

  const long long m0 = ((long long)blockIdx.x * 8 + w) * 16;
  const long long n0 = (long long)blockIdx.y * 16;

  const TA* arow = A + (m0 + lr) * lda;
  const TB* brow = B + (n0 + lr) * ldb;

  v8f acc = {0.f, 0.f, 0.f, 0.f, 0.f, 0.f, 0.f, 0.f};
  for (int k0 = 0; k0 < Kd; k0 += 32) {
    v16h av = load_a_frag(arow, k0, lg);
    v16h bv = load_b_frag(brow, k0, lg);
    acc = wmma_f16(av, bv, acc);
  }

#pragma unroll
  for (int r = 0; r < 8; ++r) {
    C[(m0 + r + lg * 8) * ldc + (n0 + lr)] = (TC)(acc[r] * scale);
  }
}

// ---------------------------------------------------------------------------
// TDM: DMA a 16-row x 64-half f16 tile (row stride 1024 halves) into LDS,
// padded to a 68-half LDS row stride (pad 2 DWORDs every 32 DWORDs) so the
// 16 lane rows map to distinct banks. D# per CDNA5 ISA.
// ---------------------------------------------------------------------------
#define QS_LD 68  // halves per LDS row after TDM padding

#if defined(__gfx1250__) && __has_builtin(__builtin_amdgcn_tensor_load_to_lds)
#define USE_TDM 1
#else
#define USE_TDM 0
#endif

__device__ __forceinline__ void tdm_load_q_tile(_Float16* lds_dst,
                                                const _Float16* gsrc) {
#if USE_TDM
  // LDS byte offset of the destination (addrspace(3) ptrtoint).
  unsigned lds_off = (unsigned)(unsigned long)(
      (__attribute__((address_space(3))) char*)lds_dst);
  unsigned long long ga = (unsigned long long)(uintptr_t)gsrc;

  v4u g0;
  g0[0] = 1u;                                   // count=1, user descriptor
  g0[1] = lds_off;                              // lds_addr (bytes)
  g0[2] = (unsigned)(ga & 0xffffffffu);         // global_addr[31:0]
  g0[3] = (unsigned)((ga >> 32) & 0x01ffffffu)  // global_addr[56:32]
          | (2u << 30);                         // type = 2 ("image")

  v8i g1;
  g1[0] = (int)((1u << 16)       // data_size = 1 -> 2-byte elements
                | (1u << 20)     // pad_enable
                | (4u << 22)     // pad_interval: 32 DWORDs (one 128B row)
                | (1u << 25));   // pad_amount: 2 DWORDs (-> 68-half stride)
  g1[1] = (int)(64u << 16);      // tensor_dim0 = 64 (low16 into [31:16])
  g1[2] = (int)(16u << 16);      // tensor_dim0 hi=0 | tensor_dim1 = 16
  g1[3] = (int)(64u << 16);      // tensor_dim1 hi=0 | tile_dim0 = 64
  g1[4] = (int)16u;              // tile_dim1 = 16, tile_dim2 = 0
  g1[5] = (int)1024u;            // tensor_dim0_stride = 1024 elements
  g1[6] = 0;                     // stride hi | tensor_dim1_stride lo
  g1[7] = 0;

  v4i gz4 = {0, 0, 0, 0};
  v8i gz8 = {0, 0, 0, 0, 0, 0, 0, 0};
  // amdgpu-toolchain (clang-23) 6-arg form:
  //   (uint32x4 g0, int32x8 g1, int32x4, int32x4, int32x8, i32 cpol)
  __builtin_amdgcn_tensor_load_to_lds(g0, g1, gz4, gz4, gz8, 0);
#if __has_builtin(__builtin_amdgcn_s_wait_tensorcnt)
  __builtin_amdgcn_s_wait_tensorcnt(0);
#endif
#else
  // Fallback: cooperative copy by one wave (32 lanes, one half each per pass).
  for (int e = threadIdx.x & 31; e < 16 * 64; e += 32) {
    int r = e >> 6, c = e & 63;
    lds_dst[r * QS_LD + c] = gsrc[r * 1024 + c];
  }
#endif
}

// ---------------------------------------------------------------------------
// Fused attention: for one (b, h) and 16 query rows, compute the full
// 16 x 2048 score panel in LDS (f32), softmax it in LDS, write attn to HBM
// exactly once, then Yh(16x64) = P(16x2048) @ Vt^T chunks, all with WMMA.
// Block = 256 threads = 8 waves. LDS ~138 KB -> 2 blocks / WGP (320 KB).
// ---------------------------------------------------------------------------
#define LKC 2048
#define SF_LD (LKC + 4)  // pad: stride 2052 words -> conflict-free col access

__global__ __launch_bounds__(256) void fused_attn_kernel(
    const _Float16* __restrict__ Qh,  // (B*LQ, 1024), head-major columns
    const _Float16* __restrict__ Kh,  // (B*LK, 1024)
    const _Float16* __restrict__ Vt,  // (B, 1024, LK): row b*1024 + h*64 + d
    float* __restrict__ attn,         // (B, H, LQ, LK)
    _Float16* __restrict__ Y1,        // (B*LQ, 1024)
    float scale) {
  __shared__ float Sf[16][SF_LD];        // score / probability panel
  __shared__ float Red[4][16][16];       // AV cross-wave K reduction
  __shared__ _Float16 Qs[16 * QS_LD];    // TDM-staged Q tile

  const int tid = threadIdx.x;
  const int w = tid >> 5;
  const int lane = tid & 31;
  const int lr = lane & 15;
  const int lg = lane >> 4;

  const int z = blockIdx.z;  // b*16 + h
  const int b = z >> 4;
  const int h = z & 15;
  const long long q0 = (long long)blockIdx.x * 16;

  const _Float16* Qbase = Qh + ((long long)b * 2048 + q0) * 1024 + h * 64;
  const _Float16* Kbase = Kh + ((long long)b * 2048) * 1024 + h * 64;
  const _Float16* Vbase = Vt + ((long long)b * 1024 + h * 64) * 2048;

  // --- Stage Q tile into LDS via the Tensor Data Mover (wave 0 issues) ----
  if (w == 0) tdm_load_q_tile(Qs, Qbase);
  __syncthreads();

  // --- Q fragments for d-steps {0,32}; reused across all 128 n-tiles ------
  v16h aq[2];
#pragma unroll
  for (int t = 0; t < 2; ++t) {
    const _Float16* qp = &Qs[lr * QS_LD + t * 32 + lg * 8];
#pragma unroll
    for (int i = 0; i < 8; ++i) aq[t][i] = qp[i];
#pragma unroll
    for (int i = 0; i < 8; ++i) aq[t][8 + i] = qp[16 + i];
  }

  // --- Scores: each wave covers n-tiles w, w+8, ... ------------------------
  for (int nt = w; nt < LKC / 16; nt += 8) {
    const int lk0 = nt * 16;
    const _Float16* krow = Kbase + (long long)(lk0 + lr) * 1024;
    // Speculative prefetch of the next tile's row (unconditional: harmless
    // over-read lands in adjacent mapped workspace; dropped if untranslated).
    __builtin_prefetch(krow + 128 * 1024, 0, 3);
    v8f acc = {0.f, 0.f, 0.f, 0.f, 0.f, 0.f, 0.f, 0.f};
#pragma unroll
    for (int t = 0; t < 2; ++t) {
      v16h bv = load_b_frag(krow, t * 32, lg);
      acc = wmma_f16(aq[t], bv, acc);
    }
#pragma unroll
    for (int r = 0; r < 8; ++r) Sf[r + lg * 8][lk0 + lr] = acc[r] * scale;
  }
  __syncthreads();

  // --- Softmax in LDS; write attn to HBM once (2 rows per wave) -----------
#pragma unroll
  for (int rr = 0; rr < 2; ++rr) {
    const int row = w * 2 + rr;
    float* p = Sf[row];

    float m = -3.402823466e38f;
    for (int c = lane; c < LKC; c += 32) m = fmaxf(m, p[c]);
#pragma unroll
    for (int off = 16; off > 0; off >>= 1) m = fmaxf(m, __shfl_xor(m, off, 32));

    float s = 0.f;
    for (int c = lane; c < LKC; c += 32) {
      float e = __expf(p[c] - m);
      p[c] = e;
      s += e;
    }
#pragma unroll
    for (int off = 16; off > 0; off >>= 1) s += __shfl_xor(s, off, 32);
    const float inv = 1.0f / s;

    float* gout = attn + ((long long)z * 2048 + q0 + row) * (long long)LKC;
    for (int c = lane; c < LKC; c += 32) {
      float pv = p[c] * inv;
      p[c] = pv;
      gout[c] = pv;
    }
  }
  __syncthreads();

  // --- Yh = P @ V: 4 d-tiles x 2 K-halves across 8 waves ------------------
  const int t = w & 3;   // d-tile (n0 = t*16)
  const int kh = w >> 2; // K half: [kh*1024, kh*1024 + 1024)
  const int n0 = t * 16;

  v8f acc = {0.f, 0.f, 0.f, 0.f, 0.f, 0.f, 0.f, 0.f};
  const _Float16* vrow = Vbase + (long long)(n0 + lr) * 2048;
  for (int k0 = kh * 1024; k0 < kh * 1024 + 1024; k0 += 32) {
    __builtin_prefetch(vrow + k0 + 32 + lg * 16, 0, 3);
    v16h av;
#pragma unroll
    for (int i = 0; i < 16; ++i) {
      int kk = k0 + ((i >> 3) << 4) + (lg << 3) + (i & 7);
      av[i] = (_Float16)Sf[lr][kk];
    }
    v16h bv = load_b_frag(vrow, k0, lg);
    acc = wmma_f16(av, bv, acc);
  }

  if (kh == 1) {
#pragma unroll
    for (int r = 0; r < 8; ++r) Red[t][r + lg * 8][lr] = acc[r];
  }
  __syncthreads();
  if (kh == 0) {
    _Float16* yout = Y1 + ((long long)b * 2048 + q0) * 1024 + h * 64;
#pragma unroll
    for (int r = 0; r < 8; ++r) {
      float v = acc[r] + Red[t][r + lg * 8][lr];
      yout[(long long)(r + lg * 8) * 1024 + n0 + lr] = (_Float16)v;
    }
  }
}

// ---------------------------------------------------------------------------
extern "C" void kernel_launch(void* const* d_in, const int* in_sizes, int n_in,
                              void* d_out, int out_size, void* d_ws,
                              size_t ws_size, hipStream_t stream) {
  constexpr int B = 2, LQ = 2048, LK = 2048;
  constexpr int KDIM = 1024, VDIM = 1024;
  constexpr int H = 16, DK = 64, DV = 64;
  constexpr long long ML = (long long)B * LQ;  // 4096

  const float* q   = (const float*)d_in[0];
  const float* k   = (const float*)d_in[1];
  const float* v   = (const float*)d_in[2];
  const float* w_q = (const float*)d_in[3];
  const float* w_k = (const float*)d_in[4];
  const float* w_v = (const float*)d_in[5];
  const float* fcy = (const float*)d_in[6];

  float* y = (float*)d_out;                    // (B, LQ, VDIM)
  float* attn = y + (long long)B * LQ * VDIM;  // (B, H, LQ, LK)

  _Float16* Qh = (_Float16*)d_ws;              // (B*LQ, 1024)
  _Float16* Kh = Qh + ML * (long long)(H * DK);
  _Float16* Vt = Kh + ML * (long long)(H * DK);      // (B, 1024, LK)
  _Float16* Y1 = Vt + (long long)B * (H * DV) * LK;  // (B*LQ, 1024)

  const dim3 blk(256);

  // Q/K projections: Xh(4096x1024) = X @ W^T  -> f16, head-major columns
  {
    dim3 g(ML / 128, (H * DK) / 16, 1);
    wmma_gemm_abt<float, float, _Float16><<<g, blk, 0, stream>>>(
        q, w_q, Qh, KDIM, (long long)KDIM, (long long)KDIM,
        (long long)(H * DK), 0, 0, 0, 1.0f);
    wmma_gemm_abt<float, float, _Float16><<<g, blk, 0, stream>>>(
        k, w_k, Kh, KDIM, (long long)KDIM, (long long)KDIM,
        (long long)(H * DK), 0, 0, 0, 1.0f);
  }
  // V projection, produced TRANSPOSED: Vt[b] (1024 x 2048) = W_v @ v[b]^T
  {
    dim3 g((H * DV) / 128, LK / 16, B);
    wmma_gemm_abt<float, float, _Float16><<<g, blk, 0, stream>>>(
        w_v, v, Vt, VDIM, (long long)VDIM, (long long)VDIM, (long long)LK,
        0, (long long)LK * VDIM, (long long)(H * DV) * LK, 1.0f);
  }

  // Fused scores + softmax + AV. attn written to HBM exactly once.
  {
    dim3 g(LQ / 16, 1, B * H);
    const float scl = rsqrtf((float)LK);  // reference scales by sqrt(len_k)
    fused_attn_kernel<<<g, blk, 0, stream>>>(Qh, Kh, Vt, attn, Y1, scl);
  }

  // y = Y1(4096x1024 f16) @ fc_y(1024x1024 f32)^T -> f32
  {
    dim3 g(ML / 128, VDIM / 16, 1);
    wmma_gemm_abt<_Float16, float, float><<<g, blk, 0, stream>>>(
        Y1, fcy, y, H * DV, (long long)(H * DV), (long long)(H * DV),
        (long long)VDIM, 0, 0, 0, 1.0f);
  }
}
